// ConversationGraphNN_83940840833693
// MI455X (gfx1250) — matrix operand order repaired and checked by reading, not compiled, not run on hardware
//
#include <hip/hip_runtime.h>
#include <hip/hip_bf16.h>
#include <math.h>

#define N_NODES 50000
#define N_EDGES 800000
#define E_TOT   (N_EDGES + N_NODES)   // self-loops appended, PyG style
#define HID     256
#define NEG_SLOPE 0.2f

typedef __attribute__((ext_vector_type(16))) __bf16        v16bf;
typedef __attribute__((ext_vector_type(8)))  float         v8f;
typedef __attribute__((ext_vector_type(8)))  unsigned int  v8u;

// ---------------- float atomic max via ordered-int trick (init must be -inf) --
__device__ inline void atomicMaxF(float* addr, float val) {
  int iv = __float_as_int(val);
  if (iv >= 0) atomicMax((int*)addr, iv);
  else         atomicMin((unsigned int*)addr, (unsigned int)iv);
}

// ---------------- generic fill ----------------------------------------------
__global__ void fill_f32(float* __restrict__ p, float v, long n) {
  long t = (long)blockIdx.x * blockDim.x + threadIdx.x;
  if (t < n) p[t] = v;
}

// ---------------- repack W [K,256] f32 -> bf16 in WMMA-B per-lane layout -----
// index = ((kt*16 + ntile)*32 + lane)*8 dwords; lane<16 holds K=kt*32+0..15,
// lane>=16 holds K=kt*32+16..31; dword v packs (K=2v) low / (K=2v+1) high.
__global__ void pack_w_bf16(const float* __restrict__ W,
                            unsigned int* __restrict__ Bpk, int K) {
  int t = blockIdx.x * blockDim.x + threadIdx.x;
  int total = (K >> 5) * 16 * 32;
  if (t >= total) return;
  int lane = t & 31;
  int nt   = (t >> 5) & 15;
  int kt   = t >> 9;
  int col  = nt * 16 + (lane & 15);
  int krow = kt * 32 + (lane >> 4) * 16;
  unsigned int* op = Bpk + (size_t)t * 8;
#pragma unroll
  for (int v = 0; v < 8; ++v) {
    __bf16 b0 = (__bf16)W[(size_t)(krow + 2 * v    ) * HID + col];
    __bf16 b1 = (__bf16)W[(size_t)(krow + 2 * v + 1) * HID + col];
    unsigned int lo = (unsigned int)__builtin_bit_cast(unsigned short, b0);
    unsigned int hi = (unsigned int)__builtin_bit_cast(unsigned short, b1);
    op[v] = (hi << 16) | lo;
  }
}

// ---------------- WMMA GEMM: C[M,256] = A[M,K] @ W[K,256], bf16 in / f32 acc -
// One wave per 16-row strip x 8 column tiles. The whole A strip (<=64 VGPRs of
// bf16) is loaded/converted ONCE into registers, then reused across all column
// tiles; only the tiny L2-resident packed B is re-read. grid = (M/16, 2).
template <int K>
__global__ void gemm_wmma_bf16(const float* __restrict__ A,
                               const unsigned int* __restrict__ Bpk,
                               float* __restrict__ C) {
  constexpr int NKT = K >> 5;           // k-steps of 32
  int mtile = blockIdx.x;
  int nt0   = blockIdx.y * 8;           // this wave covers column tiles nt0..nt0+7
  int lane  = threadIdx.x;
  int row   = mtile * 16 + (lane & 15);
  int kbase = (lane >> 4) * 16;         // lanes 0-15: K 0..15, 16-31: 16..31

  // Preload the entire A strip for this wave (f32 -> bf16, WMMA A layout)
  v16bf areg[NKT];
  const float* arow = A + (size_t)row * K;
#pragma unroll
  for (int kt = 0; kt < NKT; ++kt) {
    const float* ap = arow + kt * 32 + kbase;
#pragma unroll
    for (int i = 0; i < 8; ++i) {
      areg[kt][i]     = (__bf16)ap[i];
      areg[kt][8 + i] = (__bf16)ap[8 + i];
    }
  }

  int mrow = mtile * 16 + (lane >> 4) * 8;   // D: VGPR v -> M = v + 8*(lane>=16)
#pragma unroll
  for (int n = 0; n < 8; ++n) {
    int ntile = nt0 + n;
    v8f acc = {0.f, 0.f, 0.f, 0.f, 0.f, 0.f, 0.f, 0.f};
#pragma unroll
    for (int kt = 0; kt < NKT; ++kt) {
      const unsigned int* bp = Bpk + (((size_t)kt * 16 + ntile) * 32 + lane) * 8;
      v8u braw = *(const v8u*)bp;
      v16bf b = __builtin_bit_cast(v16bf, braw);
      acc = __builtin_amdgcn_wmma_f32_16x16x32_bf16(
          /*neg_a=*/false, areg[kt], /*neg_b=*/false, b,
          /*c_mod=*/(short)0, acc, /*reuse_a=*/false, /*reuse_b=*/false);
    }
    int ncol = ntile * 16 + (lane & 15);
#pragma unroll
    for (int v = 0; v < 8; ++v)
      C[(size_t)(mrow + v) * HID + ncol] = acc[v];
  }
}

// ---------------- per-node attention coefficients a_s, a_d -------------------
__global__ void attn_coef(const float* __restrict__ h,
                          const float* __restrict__ asrc,
                          const float* __restrict__ adst,
                          float* __restrict__ as_, float* __restrict__ ad_,
                          int H, int C) {
  int t = blockIdx.x * blockDim.x + threadIdx.x;
  if (t >= N_NODES * H) return;
  int n = t / H, hh = t % H;
  const float* hp = h + (size_t)n * HID + hh * C;
  float s = 0.f, d = 0.f;
  for (int c = 0; c < C; ++c) {
    float v = hp[c];
    s += v * asrc[hh * C + c];
    d += v * adst[hh * C + c];
  }
  as_[t] = s;
  ad_[t] = d;
}

// ---------------- edge pass 1: segment max of leaky_relu(e) ------------------
__global__ void edge_leaky_max(const int* __restrict__ ei,
                               const float* __restrict__ as_,
                               const float* __restrict__ ad_,
                               float* __restrict__ m, int H, int hshift) {
  long t = (long)blockIdx.x * blockDim.x + threadIdx.x;
  long n = (long)E_TOT * H;
  if (t >= n) return;
  int e  = (int)(t >> hshift);
  int hh = (int)(t & (H - 1));
  int s, d;
  if (e < N_EDGES) { s = ei[e]; d = ei[N_EDGES + e]; }
  else             { s = d = e - N_EDGES; }
  float ev = as_[s * H + hh] + ad_[d * H + hh];
  ev = ev > 0.f ? ev : NEG_SLOPE * ev;
  atomicMaxF(&m[d * H + hh], ev);
}

// ---------------- edge pass 2: segment sum of exp(e - m[dst]) ----------------
__global__ void edge_exp_sum(const int* __restrict__ ei,
                             const float* __restrict__ as_,
                             const float* __restrict__ ad_,
                             const float* __restrict__ m,
                             float* __restrict__ den, int H, int hshift) {
  long t = (long)blockIdx.x * blockDim.x + threadIdx.x;
  long n = (long)E_TOT * H;
  if (t >= n) return;
  int e  = (int)(t >> hshift);
  int hh = (int)(t & (H - 1));
  int s, d;
  if (e < N_EDGES) { s = ei[e]; d = ei[N_EDGES + e]; }
  else             { s = d = e - N_EDGES; }
  int idx = d * H + hh;
  float ev = as_[s * H + hh] + ad_[idx];
  ev = ev > 0.f ? ev : NEG_SLOPE * ev;
  atomicAdd(&den[idx], __expf(ev - m[idx]));
}

// ---------------- edge pass 3: out[dst] += h[src] * alpha --------------------
// one thread per (edge, channel); alpha recomputed (cheaper than storing it)
__global__ void edge_message(const int* __restrict__ ei,
                             const float* __restrict__ as_,
                             const float* __restrict__ ad_,
                             const float* __restrict__ m,
                             const float* __restrict__ den,
                             const float* __restrict__ h,
                             float* __restrict__ out, int H, int cshift) {
  long t = (long)blockIdx.x * blockDim.x + threadIdx.x;
  long n = (long)E_TOT * HID;
  if (t >= n) return;
  int e  = (int)(t >> 8);        // HID == 256
  int j  = (int)(t & 255);
  int hh = j >> cshift;
  int s, d;
  if (e < N_EDGES) { s = ei[e]; d = ei[N_EDGES + e]; }
  else             { s = d = e - N_EDGES; }
  int idx = d * H + hh;
  float ev = as_[s * H + hh] + ad_[idx];
  ev = ev > 0.f ? ev : NEG_SLOPE * ev;
  float alpha = __expf(ev - m[idx]) / den[idx];
  atomicAdd(&out[(size_t)d * HID + j], h[(size_t)s * HID + j] * alpha);
}

// ---------------- bias add (+ optional ELU) ----------------------------------
__global__ void bias_act(const float* __restrict__ in, const float* __restrict__ b,
                         float* __restrict__ out, int do_elu) {
  size_t t = (size_t)blockIdx.x * blockDim.x + threadIdx.x;
  if (t >= (size_t)N_NODES * HID) return;
  float v = in[t] + b[t & 255];
  if (do_elu) v = v > 0.f ? v : (__expf(v) - 1.f);
  out[t] = v;
}

// ---------------- mean pool over nodes: one block per column -----------------
__global__ void col_mean(const float* __restrict__ h, float* __restrict__ g) {
  __shared__ float red[256];
  int j = blockIdx.x;
  float s = 0.f;
  for (int i = threadIdx.x; i < N_NODES; i += 256) s += h[(size_t)i * HID + j];
  red[threadIdx.x] = s;
  __syncthreads();
  for (int w = 128; w > 0; w >>= 1) {
    if (threadIdx.x < w) red[threadIdx.x] += red[threadIdx.x + w];
    __syncthreads();
  }
  if (threadIdx.x == 0) g[j] = red[0] / (float)N_NODES;
}

// ---------------- tiny MLP heads (single block) ------------------------------
__global__ void mlp_heads(const float* __restrict__ g,
                          const float* ow1, const float* ob1,
                          const float* ow2, const float* ob2,
                          const float* ew1, const float* eb1,
                          const float* ew2, const float* eb2,
                          const float* dw1, const float* db1,
                          const float* dw2, const float* db2,
                          float* __restrict__ out) {
  __shared__ float gs[HID];
  __shared__ float z1[128], ze[64], zd[64];
  int tid = threadIdx.x;
  gs[tid] = g[tid];
  __syncthreads();
  if (tid < 128) {
    float s = ob1[tid];
    for (int k = 0; k < HID; ++k) s += gs[k] * ow1[k * 128 + tid];
    z1[tid] = fmaxf(s, 0.f);
  }
  if (tid < 64) {
    float s = eb1[tid];
    for (int k = 0; k < HID; ++k) s += gs[k] * ew1[k * 64 + tid];
    ze[tid] = fmaxf(s, 0.f);
    float s2 = db1[tid];
    for (int k = 0; k < HID; ++k) s2 += gs[k] * dw1[k * 64 + tid];
    zd[tid] = fmaxf(s2, 0.f);
  }
  __syncthreads();
  if (tid < 6) {
    float s = ob2[tid];
    for (int k = 0; k < 128; ++k) s += z1[k] * ow2[k * 6 + tid];
    out[tid] = 1.f / (1.f + __expf(-s));
  }
  if (tid == 6) {
    float s = eb2[0];
    for (int k = 0; k < 64; ++k) s += ze[k] * ew2[k];
    out[6] = 1.f / (1.f + __expf(-s));
  }
  if (tid == 7) {
    float s = db2[0];
    for (int k = 0; k < 64; ++k) s += zd[k] * dw2[k];
    out[7] = 1.f / (1.f + __expf(-s));
  }
}

// =============================================================================
extern "C" void kernel_launch(void* const* d_in, const int* in_sizes, int n_in,
                              void* d_out, int out_size, void* d_ws, size_t ws_size,
                              hipStream_t stream) {
  (void)in_sizes; (void)n_in; (void)out_size; (void)ws_size;
  const float* x  = (const float*)d_in[0];
  const int*   ei = (const int*)d_in[1];
  const float* Wm[3]   = {(const float*)d_in[2],  (const float*)d_in[6],  (const float*)d_in[10]};
  const float* Asrc[3] = {(const float*)d_in[3],  (const float*)d_in[7],  (const float*)d_in[11]};
  const float* Adst[3] = {(const float*)d_in[4],  (const float*)d_in[8],  (const float*)d_in[12]};
  const float* Bias[3] = {(const float*)d_in[5],  (const float*)d_in[9],  (const float*)d_in[13]};
  const int Hs[3] = {4, 4, 1}, Cs[3] = {64, 64, 256}, Ks[3] = {128, 256, 256};
  const int cshift[3] = {6, 6, 8}, hshift[3] = {2, 2, 0};

  float* out = (float*)d_out;
  const size_t NH = (size_t)N_NODES * HID;

  char* ws = (char*)d_ws;
  float* hproj = (float*)ws;  ws += NH * sizeof(float);
  float* hin   = (float*)ws;  ws += NH * sizeof(float);
  float* hagg  = (float*)ws;  ws += NH * sizeof(float);
  float* asb   = (float*)ws;  ws += (size_t)N_NODES * 4 * sizeof(float);
  float* adb   = (float*)ws;  ws += (size_t)N_NODES * 4 * sizeof(float);
  float* mb    = (float*)ws;  ws += (size_t)N_NODES * 4 * sizeof(float);
  float* db    = (float*)ws;  ws += (size_t)N_NODES * 4 * sizeof(float);
  unsigned int* wpk = (unsigned int*)ws;   // up to (256/32)*16*32*8 dwords = 128 KB

  for (int i = 0; i < 3; ++i) {
    const float* in = (i == 0) ? x : hin;
    const int K = Ks[i], H = Hs[i], C = Cs[i];

    int pt = (K >> 5) * 16 * 32;
    pack_w_bf16<<<(pt + 255) / 256, 256, 0, stream>>>(Wm[i], wpk, K);

    dim3 gg(N_NODES / 16, 2);           // each wave: 16-row strip x 8 col tiles
    if (K == 128) gemm_wmma_bf16<128><<<gg, 32, 0, stream>>>(in, wpk, hproj);
    else          gemm_wmma_bf16<256><<<gg, 32, 0, stream>>>(in, wpk, hproj);

    int NHv = N_NODES * H;
    attn_coef<<<(NHv + 255) / 256, 256, 0, stream>>>(hproj, Asrc[i], Adst[i], asb, adb, H, C);

    fill_f32<<<(NHv + 255) / 256, 256, 0, stream>>>(mb, -INFINITY, (long)NHv);
    fill_f32<<<(NHv + 255) / 256, 256, 0, stream>>>(db, 0.f, (long)NHv);
    fill_f32<<<(int)((NH + 255) / 256), 256, 0, stream>>>(hagg, 0.f, (long)NH);

    long ne = (long)E_TOT * H;
    edge_leaky_max<<<(int)((ne + 255) / 256), 256, 0, stream>>>(ei, asb, adb, mb, H, hshift[i]);
    edge_exp_sum  <<<(int)((ne + 255) / 256), 256, 0, stream>>>(ei, asb, adb, mb, db, H, hshift[i]);

    long nm = (long)E_TOT * HID;
    edge_message<<<(int)((nm + 255) / 256), 256, 0, stream>>>(ei, asb, adb, mb, db, hproj, hagg, H, cshift[i]);

    float* dst = (i == 2) ? out : hin;
    bias_act<<<(int)((NH + 255) / 256), 256, 0, stream>>>(hagg, Bias[i], dst, (i < 2) ? 1 : 0);
  }

  float* gr = out + NH;               // graph_repr [256]
  col_mean<<<HID, 256, 0, stream>>>(out, gr);
  mlp_heads<<<1, 256, 0, stream>>>(gr,
      (const float*)d_in[14], (const float*)d_in[15], (const float*)d_in[16], (const float*)d_in[17],
      (const float*)d_in[18], (const float*)d_in[19], (const float*)d_in[20], (const float*)d_in[21],
      (const float*)d_in[22], (const float*)d_in[23], (const float*)d_in[24], (const float*)d_in[25],
      gr + HID);                      // outcomes[6], engagement[1], difficulty[1]
}
